// Transformer_54305566491161
// MI455X (gfx1250) — compile-verified, matrix operand order checked by reading, not run on hardware
//
#include <hip/hip_runtime.h>
#include <hip/hip_bf16.h>

typedef __attribute__((ext_vector_type(16))) _Float16 v16h;
typedef __attribute__((ext_vector_type(8)))  _Float16 v8h;
typedef __attribute__((ext_vector_type(8)))  float    v8f;

constexpr int kE    = 384;
constexpr int kH    = 6;
constexpr int kHS   = 64;
constexpr int kT    = 256;
constexpr int kB    = 16;
constexpr int kL    = 6;
constexpr int kV    = 50257;
constexpr int kRows = kB * kT;      // 4096
constexpr int kNpad = 50304;        // kV padded to multiple of 128

// ---------------------------------------------------------------- utilities
__device__ inline float wave_max(float v) {
#pragma unroll
  for (int o = 16; o > 0; o >>= 1) v = fmaxf(v, __shfl_xor(v, o, 32));
  return v;
}
__device__ inline float wave_sum(float v) {
#pragma unroll
  for (int o = 16; o > 0; o >>= 1) v += __shfl_xor(v, o, 32);
  return v;
}

// Load one 16x32 f16 WMMA operand fragment (A row-major [M,K] or B as [N,K]).
// Lanes 0-15: rows 0-15, K=kb..kb+7 (VGPR0-3) and K=16+kb..16+kb+7 (VGPR4-7),
// kb=0; lanes 16-31: same rows, kb=8. Two contiguous 16-byte loads per lane.
__device__ inline v16h load_frag(const _Float16* __restrict__ p, int ld,
                                 int row, int k0, int lane) {
  int r  = row + (lane & 15);
  int kb = (lane >> 4) << 3;
  const _Float16* base = p + (size_t)r * ld + k0 + kb;
  v8h lo = *(const v8h*)(base);
  v8h hi = *(const v8h*)(base + 16);
  v16h f;
#pragma unroll
  for (int i = 0; i < 8; ++i) { f[i] = lo[i]; f[i + 8] = hi[i]; }
  return f;
}

// -------------------------------------------------------------- WMMA GEMM
// C[m,n] = sum_k A[m,k] * B[n,k]   (B stored transposed, [N,K] row-major)
// Wave tile: 64x64 (4x4 16x16 accumulators; 8 fragment loads -> 16 WMMAs).
// Block tile: (WM*64) x (WN*64), WM*WN waves.
// EPI: 0=none, 1=+bias, 2=+bias+residual(C), 3=relu(acc+bias)+residual(C)
// OUTH: write f16 to Ch instead of f32 to C.  BCHECK: guard n < Nreal.
// Batch z offsets: off = (z/zdiv)*zXo + (z%zdiv)*zXi for A, B, C.
template <int WM, int WN, int EPI, bool OUTH, bool BCHECK>
__global__ void gemm_wmma(const _Float16* __restrict__ A, int lda,
                          const _Float16* __restrict__ Bm, int ldb,
                          float* C, _Float16* Ch, int ldc,
                          const float* __restrict__ bias,
                          int Nreal, int K, int zdiv,
                          long zAo, long zAi, long zBo, long zBi,
                          long zCo, long zCi) {
  int z  = blockIdx.z;
  int zo = z / zdiv, zi = z - zo * zdiv;
  A  += zo * zAo + zi * zAi;
  Bm += zo * zBo + zi * zBi;
  long cOff = zo * zCo + zi * zCi;

  int lane = threadIdx.x & 31;
  int wave = threadIdx.x >> 5;
  int wm = wave % WM, wn = wave / WM;
  int rowBase = blockIdx.y * (WM * 64) + wm * 64;
  int colBase = blockIdx.x * (WN * 64) + wn * 64;

  v8f acc[4][4];
#pragma unroll
  for (int i = 0; i < 4; ++i)
#pragma unroll
    for (int j = 0; j < 4; ++j) acc[i][j] = 0.f;

  for (int k0 = 0; k0 < K; k0 += 32) {
    v16h a[4], b[4];
#pragma unroll
    for (int i = 0; i < 4; ++i) a[i] = load_frag(A,  lda, rowBase + 16 * i, k0, lane);
#pragma unroll
    for (int j = 0; j < 4; ++j) b[j] = load_frag(Bm, ldb, colBase + 16 * j, k0, lane);
#pragma unroll
    for (int i = 0; i < 4; ++i)
#pragma unroll
      for (int j = 0; j < 4; ++j)
        acc[i][j] = __builtin_amdgcn_wmma_f32_16x16x32_f16(
            false, a[i], false, b[j], (short)0, acc[i][j], false, false);
  }

  int nl   = lane & 15;
  int rofs = (lane >> 4) << 3;   // lanes 16-31 hold rows M=8..15 of each tile
#pragma unroll
  for (int i = 0; i < 4; ++i) {
#pragma unroll
    for (int j = 0; j < 4; ++j) {
      int n = colBase + j * 16 + nl;
      if (BCHECK && n >= Nreal) continue;
      float bv = (EPI >= 1) ? bias[n] : 0.f;
      long base = cOff + (long)(rowBase + i * 16 + rofs) * ldc + n;
#pragma unroll
      for (int r = 0; r < 8; ++r) {
        long idx = base + (long)r * ldc;
        float v = acc[i][j][r];
        if (EPI == 1)      v += bv;
        else if (EPI == 2) v += bv + C[idx];
        else if (EPI == 3) v = fmaxf(v + bv, 0.f) + C[idx];
        if (OUTH) Ch[idx] = (_Float16)v; else C[idx] = v;
      }
    }
  }
}

// ------------------------------------------------- weight convert+transpose
// W: nmat stacked [K,N] f32  ->  Wt: nmat stacked [Npad,K] f16 (rows>=N zero)
__global__ void convert_wT(const float* __restrict__ W, _Float16* __restrict__ Wt,
                           int K, int N, int Npad, int nmat) {
  long total = (long)nmat * Npad * K;
  for (long i = (long)blockIdx.x * blockDim.x + threadIdx.x; i < total;
       i += (long)gridDim.x * blockDim.x) {
    int  k   = (int)(i % K);
    long t   = i / K;
    int  n   = (int)(t % Npad);
    int  mat = (int)(t / Npad);
    float v = (n < N) ? W[((long)mat * K + k) * N + n] : 0.f;
    Wt[i] = (_Float16)v;
  }
}

// ------------------------------------------------------------------- embed
__global__ void embed_kernel(const int* __restrict__ idx,
                             const float* __restrict__ tok,
                             const float* __restrict__ pos,
                             float* __restrict__ x) {
  int row = blockIdx.x;                 // 0..4095
  int t   = row & (kT - 1);
  long tb = (long)idx[row] * kE;
  for (int e = threadIdx.x; e < kE; e += blockDim.x)
    x[(long)row * kE + e] = tok[tb + e] + pos[(long)t * kE + e];
}

// --------------------------------------------------------------- layernorm
// One wave per row; f32 in, f16 out (GEMM operand).
__global__ void layernorm_kernel(const float* __restrict__ x,
                                 const float* __restrict__ g,
                                 const float* __restrict__ b,
                                 _Float16* __restrict__ out, int rows) {
  int row = blockIdx.x * (blockDim.x >> 5) + (threadIdx.x >> 5);
  if (row >= rows) return;
  int lane = threadIdx.x & 31;
  const float* xr = x + (long)row * kE;
  float v[12];
  float s = 0.f, sq = 0.f;
#pragma unroll
  for (int i = 0; i < 3; ++i) {
#pragma unroll
    for (int c = 0; c < 4; ++c) {
      float f = xr[lane * 4 + i * 128 + c];
      v[i * 4 + c] = f; s += f; sq += f * f;
    }
  }
  s  = wave_sum(s);
  sq = wave_sum(sq);
  float mu   = s / kE;
  float var  = sq / kE - mu * mu;       // biased variance (matches reference)
  float rstd = rsqrtf(var + 1e-5f);
#pragma unroll
  for (int i = 0; i < 3; ++i) {
#pragma unroll
    for (int c = 0; c < 4; ++c) {
      int e = lane * 4 + i * 128 + c;
      out[(long)row * kE + e] = (_Float16)((v[i * 4 + c] - mu) * rstd * g[e] + b[e]);
    }
  }
}

// ------------------------------------------------------ causal softmax row
// S [BH,256,256] f32 -> P [BH,256,256] f16; mask k>q to 0, scale = 1/sqrt(HS)
__global__ void softmax_kernel(const float* __restrict__ S,
                               _Float16* __restrict__ P, float scale) {
  int row  = blockIdx.x * (blockDim.x >> 5) + (threadIdx.x >> 5); // 0..24575
  int lane = threadIdx.x & 31;
  int z = row >> 8, q = row & 255;
  const float* s = S + (long)z * 65536 + (long)q * 256;
  _Float16*    p = P + (long)z * 65536 + (long)q * 256;
  float vals[8];
  float mx = -1e30f;
#pragma unroll
  for (int i = 0; i < 8; ++i) {
    int k = lane * 8 + i;
    float f = (k <= q) ? s[k] * scale : -1e30f;
    vals[i] = f; mx = fmaxf(mx, f);
  }
  mx = wave_max(mx);
  float sum = 0.f;
#pragma unroll
  for (int i = 0; i < 8; ++i) {
    float e = __expf(vals[i] - mx);     // masked lanes underflow to 0
    vals[i] = e; sum += e;
  }
  sum = wave_sum(sum);
  float inv = 1.f / sum;
#pragma unroll
  for (int i = 0; i < 8; ++i) p[lane * 8 + i] = (_Float16)(vals[i] * inv);
}

// -------------------------------------------- V [B,T,H*HS] -> vT [BH,HS,T]
__global__ void transpose_v(const _Float16* __restrict__ v,
                            _Float16* __restrict__ vT) {
  long total = (long)kB * kH * kHS * kT;
  for (long i = (long)blockIdx.x * blockDim.x + threadIdx.x; i < total;
       i += (long)gridDim.x * blockDim.x) {
    int  n = (int)(i % kT);
    long t = i / kT;
    int  d = (int)(t % kHS);
    int  z = (int)(t / kHS);
    int  bb = z / kH, hh = z - bb * kH;
    vT[i] = v[((long)bb * kT + n) * kE + hh * kHS + d];
  }
}

// =========================================================================
extern "C" void kernel_launch(void* const* d_in, const int* in_sizes, int n_in,
                              void* d_out, int out_size, void* d_ws, size_t ws_size,
                              hipStream_t stream) {
  (void)in_sizes; (void)n_in; (void)out_size; (void)ws_size;
  const int*   idx  = (const int*)  d_in[0];
  const float* tokE = (const float*)d_in[1];
  const float* posE = (const float*)d_in[2];
  const float* Wq   = (const float*)d_in[3];
  const float* Wk   = (const float*)d_in[4];
  const float* Wv   = (const float*)d_in[5];
  const float* Wo   = (const float*)d_in[6];
  const float* bo   = (const float*)d_in[7];
  const float* ln1g = (const float*)d_in[8];
  const float* ln1b = (const float*)d_in[9];
  const float* ln2g = (const float*)d_in[10];
  const float* ln2b = (const float*)d_in[11];
  const float* Wm   = (const float*)d_in[12];
  const float* bm   = (const float*)d_in[13];
  const float* lnfg = (const float*)d_in[14];
  const float* lnfb = (const float*)d_in[15];
  const float* Wlm  = (const float*)d_in[16];
  const float* blm  = (const float*)d_in[17];
  float* out = (float*)d_out;

  char* ws = (char*)d_ws; size_t off = 0;
  auto alloc = [&](size_t bytes) -> void* {
    void* p = ws + off; off = (off + bytes + 255) & ~(size_t)255; return p;
  };
  float*    x    = (float*)   alloc((size_t)kRows * kE * 4);        // residual stream
  _Float16* h    = (_Float16*)alloc((size_t)kRows * kE * 2);        // LN output (f16)
  _Float16* q    = (_Float16*)alloc((size_t)kRows * kE * 2);
  _Float16* k_   = (_Float16*)alloc((size_t)kRows * kE * 2);
  _Float16* v_   = (_Float16*)alloc((size_t)kRows * kE * 2);
  _Float16* vT   = (_Float16*)alloc((size_t)kB * kH * kHS * kT * 2);
  _Float16* o    = (_Float16*)alloc((size_t)kRows * kE * 2);
  float*    S    = (float*)   alloc((size_t)kB * kH * kT * kT * 4); // 25 MB (L2-resident)
  _Float16* P    = (_Float16*)alloc((size_t)kB * kH * kT * kT * 2);
  _Float16* WqT  = (_Float16*)alloc((size_t)kL * kE * kE * 2);
  _Float16* WkT  = (_Float16*)alloc((size_t)kL * kE * kE * 2);
  _Float16* WvT  = (_Float16*)alloc((size_t)kL * kE * kE * 2);
  _Float16* WoT  = (_Float16*)alloc((size_t)kL * kE * kE * 2);
  _Float16* WmT  = (_Float16*)alloc((size_t)kL * kE * kE * 2);
  _Float16* WlmT = (_Float16*)alloc((size_t)kNpad * kE * 2);

  // ---- stage weights: f32 [K,N] -> f16 [Npad,K] (transposed for WMMA B-operand)
  convert_wT<<<2048, 256, 0, stream>>>(Wq,  WqT,  kE, kE, kE, kL);
  convert_wT<<<2048, 256, 0, stream>>>(Wk,  WkT,  kE, kE, kE, kL);
  convert_wT<<<2048, 256, 0, stream>>>(Wv,  WvT,  kE, kE, kE, kL);
  convert_wT<<<2048, 256, 0, stream>>>(Wo,  WoT,  kE, kE, kE, kL);
  convert_wT<<<2048, 256, 0, stream>>>(Wm,  WmT,  kE, kE, kE, kL);
  convert_wT<<<8192, 256, 0, stream>>>(Wlm, WlmT, kE, kV, kNpad, 1);

  // ---- embedding
  embed_kernel<<<kRows, 128, 0, stream>>>(idx, tokE, posE, x);

  dim3 gProj(kE / 128, kRows / 128, 1);               // (3, 32)
  dim3 gScore(kT / 128, kT / 128, kB * kH);           // (2, 2, 96)
  dim3 gAttO(1, kT / 128, kB * kH);                   // (1, 2, 96)
  const long sQK = (long)kT * kE;                     // 98304: per-batch q/k/v/o stride
  const long sSS = (long)kT * kT;                     // 65536: per-(b,h) score stride
  const long sVT = (long)kHS * kT;                    // 16384

  for (int l = 0; l < kL; ++l) {
    const size_t wo_ = (size_t)l * kE * kE;
    // LN1 -> h (f16)
    layernorm_kernel<<<kRows / 8, 256, 0, stream>>>(x, ln1g + l * kE, ln1b + l * kE, h, kRows);
    // q,k,v = h @ W  (no bias in reference)
    gemm_wmma<2,2,0,true,false><<<gProj, 128, 0, stream>>>(h, kE, WqT + wo_, kE,
        nullptr, q,  kE, nullptr, kE, kE, 1, 0,0, 0,0, 0,0);
    gemm_wmma<2,2,0,true,false><<<gProj, 128, 0, stream>>>(h, kE, WkT + wo_, kE,
        nullptr, k_, kE, nullptr, kE, kE, 1, 0,0, 0,0, 0,0);
    gemm_wmma<2,2,0,true,false><<<gProj, 128, 0, stream>>>(h, kE, WvT + wo_, kE,
        nullptr, v_, kE, nullptr, kE, kE, 1, 0,0, 0,0, 0,0);
    // S[z] = Q[z] K[z]^T   (z = b*H + h; A/B offsets: b*sQK + h*64)
    gemm_wmma<2,2,0,false,false><<<gScore, 128, 0, stream>>>(q, kE, k_, kE,
        S, nullptr, kT, nullptr, kT, kHS, kH,
        sQK, (long)kHS, sQK, (long)kHS, (long)kH * sSS, sSS);
    // causal softmax (mask, scale, normalize) -> P (f16)
    softmax_kernel<<<(kB * kH * kT) / 8, 256, 0, stream>>>(S, P, 0.125f);
    // V -> vT [BH, HS, T]
    transpose_v<<<3072, 256, 0, stream>>>(v_, vT);
    // O[z] = P[z] V[z]  -> o laid out as [B,T,E] (offset b*sQK + h*64)
    gemm_wmma<2,1,0,true,false><<<gAttO, 64, 0, stream>>>(P, kT, vT, kT,
        nullptr, o, kE, nullptr, kHS, kT, kH,
        (long)kH * sSS, sSS, (long)kH * sVT, sVT, sQK, (long)kHS);
    // x += o @ Wo + bo   (in-place residual)
    gemm_wmma<2,2,2,false,false><<<gProj, 128, 0, stream>>>(o, kE, WoT + wo_, kE,
        x, nullptr, kE, bo + l * kE, kE, kE, 1, 0,0, 0,0, 0,0);
    // LN2 -> h
    layernorm_kernel<<<kRows / 8, 256, 0, stream>>>(x, ln2g + l * kE, ln2b + l * kE, h, kRows);
    // x += relu(h @ Wm + bm)
    gemm_wmma<2,2,3,false,false><<<gProj, 128, 0, stream>>>(h, kE, WmT + wo_, kE,
        x, nullptr, kE, bm + l * kE, kE, kE, 1, 0,0, 0,0, 0,0);
  }

  // final LN -> h ; logits = h @ Wlm + blm (bounds-checked at N=50257)
  layernorm_kernel<<<kRows / 8, 256, 0, stream>>>(x, lnfg, lnfb, h, kRows);
  dim3 gLM(kNpad / 128, kRows / 128, 1);              // (393, 32)
  gemm_wmma<2,2,1,false,true><<<gLM, 128, 0, stream>>>(h, kE, WlmT, kE,
      out, nullptr, kV, blm, kV, kE, 1, 0,0, 0,0, 0,0);
}
